// LinearizedLowRankModel_84920093376889
// MI455X (gfx1250) — matrix operand order repaired and checked by reading, not compiled, not run on hardware
//
#include <hip/hip_runtime.h>

// ---------------------------------------------------------------------------
// Model constants (from reference)
// ---------------------------------------------------------------------------
#define L_LAYERS 2
#define BB 2
#define SS 1024
#define DD 1024
#define HH 16
#define HD 64
#define FF 4096
#define VV 32000
#define RR 16
#define ROWS (BB * SS)          // 2048
#define LORA_SCALE 2.0f
#define EPS_RMS 1e-6f
#define ATT_CHUNK 8             // heads per attention chunk (divides HH)

// ---------------------------------------------------------------------------
// WMMA GEMM: C[M,N] = alpha * A[M,K](f32) x B(f32) (+C if accumulate)
//   transB=0: B is KxN row-major (ldb).  transB=1: B is NxK row-major (ldb).
//   Batched over blockIdx.z with element strides sAz/sBz/sCz.
//   fp32 -> bf16 (RNE) conversion during LDS staging; fp32 accumulate via
//   v_wmma_f32_16x16x32_bf16.
// ---------------------------------------------------------------------------
#define BM 128
#define BN 128
#define BK 32
#define BKP 40   // padded LDS row (ushorts): 80B rows -> 16B-aligned fragments

typedef __attribute__((ext_vector_type(16))) __bf16 v16bf;
typedef __attribute__((ext_vector_type(8)))  float  v8f;

union ABFrag { v16bf v; unsigned short s[16]; };
union CFrag  { v8f v; float f[8]; };

__device__ __forceinline__ unsigned short f2bf_rne(float f) {
  unsigned int u = __float_as_uint(f);
  unsigned int r = 0x7FFFu + ((u >> 16) & 1u);
  return (unsigned short)((u + r) >> 16);
}
__device__ __forceinline__ unsigned int pack_bf2(float a, float b) {
  return (unsigned int)f2bf_rne(a) | ((unsigned int)f2bf_rne(b) << 16);
}

__global__ __launch_bounds__(256)
void gemm_bf16_wmma(const float* __restrict__ A, const float* __restrict__ B,
                    float* __restrict__ C,
                    int M, int N, int K, int lda, int ldb, int ldc,
                    long long sAz, long long sBz, long long sCz,
                    float alpha, int transB, int accumulate)
{
  const int z = blockIdx.z;
  A += (long long)z * sAz;
  B += (long long)z * sBz;
  C += (long long)z * sCz;

  const int m0   = blockIdx.y * BM;
  const int n0   = blockIdx.x * BN;
  const int tid  = threadIdx.x;
  const int lane = tid & 31;
  const int warp = tid >> 5;
  const int mw   = (warp & 3) * 32;   // 4 wave-rows
  const int nw   = (warp >> 2) * 64;  // 2 wave-cols
  const int lmod = lane & 15;
  const int kbaseA = (lane >> 4) * 8;   // A: K interleaved by 8s per half-wave
  const int kbaseB = (lane >> 4) * 8;   // B: K interleaved by 8s per half-wave

  __shared__ unsigned short As[BM][BKP];
  __shared__ unsigned short Bs[BN][BKP];   // stored N-major: Bs[n][k]

  const bool mFull = (m0 + BM <= M);
  const bool nFull = (n0 + BN <= N);

  CFrag acc[2][4];
#pragma unroll
  for (int i = 0; i < 2; ++i)
#pragma unroll
    for (int j = 0; j < 4; ++j)
#pragma unroll
      for (int r = 0; r < 8; ++r) acc[i][j].f[r] = 0.0f;

  for (int k0 = 0; k0 < K; k0 += BK) {
    const bool kFull = (k0 + BK <= K);

    // ---------------- stage A tile (row-major over k) ----------------
    if (mFull && kFull) {
      if (k0 + BK < K)
        __builtin_prefetch(&A[(size_t)(m0 + (tid >> 3)) * lda + k0 + BK], 0, 1);
#pragma unroll
      for (int p = 0; p < 4; ++p) {
        const int idx = tid + 256 * p;          // 0..1023 float4s
        const int r   = idx >> 3;               // 8 float4 per 32-float row
        const int c4  = (idx & 7) * 4;
        const float4 v = *(const float4*)&A[(size_t)(m0 + r) * lda + k0 + c4];
        uint2 w; w.x = pack_bf2(v.x, v.y); w.y = pack_bf2(v.z, v.w);
        *(uint2*)&As[r][c4] = w;                // ds_store_b64
      }
    } else {
      const int c = tid & 31;
      const int rbase = tid >> 5;
#pragma unroll
      for (int p = 0; p < 16; ++p) {
        const int r  = rbase + p * 8;
        const int gm = m0 + r, gk = k0 + c;
        float v = (gm < M && gk < K) ? A[(size_t)gm * lda + gk] : 0.0f;
        As[r][c] = f2bf_rne(v);
      }
    }

    // ---------------- stage B tile into Bs[n][k] ----------------
    if (transB) {
      if (nFull && kFull) {
        if (k0 + BK < K)
          __builtin_prefetch(&B[(size_t)(n0 + (tid >> 3)) * ldb + k0 + BK], 0, 1);
#pragma unroll
        for (int p = 0; p < 4; ++p) {
          const int idx = tid + 256 * p;
          const int n   = idx >> 3;
          const int c4  = (idx & 7) * 4;
          const float4 v = *(const float4*)&B[(size_t)(n0 + n) * ldb + k0 + c4];
          uint2 w; w.x = pack_bf2(v.x, v.y); w.y = pack_bf2(v.z, v.w);
          *(uint2*)&Bs[n][c4] = w;
        }
      } else {
        const int c = tid & 31;
        const int nb = tid >> 5;
#pragma unroll
        for (int p = 0; p < 16; ++p) {
          const int n  = nb + p * 8;
          const int gn = n0 + n, gk = k0 + c;
          float v = (gn < N && gk < K) ? B[(size_t)gn * ldb + gk] : 0.0f;
          Bs[n][c] = f2bf_rne(v);
        }
      }
    } else {
      if (nFull && kFull) {
        if (k0 + BK < K)
          __builtin_prefetch(&B[(size_t)(k0 + BK + (tid >> 5)) * ldb + n0 + (tid & 31) * 4], 0, 1);
#pragma unroll
        for (int p = 0; p < 4; ++p) {
          const int idx = tid + 256 * p;        // 0..1023 float4s
          const int k   = idx >> 5;             // 32 float4 per 128-float row
          const int n4  = (idx & 31) * 4;
          const float4 v = *(const float4*)&B[(size_t)(k0 + k) * ldb + n0 + n4];
          Bs[n4 + 0][k] = f2bf_rne(v.x);        // transpose scatter
          Bs[n4 + 1][k] = f2bf_rne(v.y);
          Bs[n4 + 2][k] = f2bf_rne(v.z);
          Bs[n4 + 3][k] = f2bf_rne(v.w);
        }
      } else {
        const int n  = tid & 127;
        const int kb = tid >> 7;
#pragma unroll
        for (int p = 0; p < 16; ++p) {
          const int kk = kb + p * 2;
          const int gn = n0 + n, gk = k0 + kk;
          float v = (gn < N && gk < K) ? B[(size_t)gk * ldb + gn] : 0.0f;
          Bs[n][kk] = f2bf_rne(v);
        }
      }
    }
    __syncthreads();

    // ---- load fragments (ISA VGPR layouts); rows 16B-aligned -> b128 ----
    ABFrag af[2], bf[4];
#pragma unroll
    for (int sm = 0; sm < 2; ++sm) {
      const int row = mw + sm * 16 + lmod;
      *(uint4*)&af[sm].s[0] = *(const uint4*)&As[row][kbaseA];
      *(uint4*)&af[sm].s[8] = *(const uint4*)&As[row][16 + kbaseA];
    }
#pragma unroll
    for (int sn = 0; sn < 4; ++sn) {
      const int col = nw + sn * 16 + lmod;
      *(uint4*)&bf[sn].s[0] = *(const uint4*)&Bs[col][kbaseB];
      *(uint4*)&bf[sn].s[8] = *(const uint4*)&Bs[col][16 + kbaseB];
    }

    // ---- 8 WMMAs per wave per K-step ----
#pragma unroll
    for (int sm = 0; sm < 2; ++sm)
#pragma unroll
      for (int sn = 0; sn < 4; ++sn)
        acc[sm][sn].v = __builtin_amdgcn_wmma_f32_16x16x32_bf16(
            false, af[sm].v, false, bf[sn].v, (short)0, acc[sm][sn].v, false, false);

    __syncthreads();
  }

  // ---- epilogue: C VGPR r <-> M = r + 8*(lane>=16), N = lane&15 ----
  const int mi = (lane >> 4) * 8;
#pragma unroll
  for (int sm = 0; sm < 2; ++sm) {
#pragma unroll
    for (int sn = 0; sn < 4; ++sn) {
#pragma unroll
      for (int r = 0; r < 8; ++r) {
        const int gm = m0 + mw + sm * 16 + mi + r;
        const int gn = n0 + nw + sn * 16 + lmod;
        if (gm < M && gn < N) {
          const size_t off = (size_t)gm * ldc + gn;
          float v = alpha * acc[sm][sn].f[r];
          if (accumulate) v += C[off];
          C[off] = v;
        }
      }
    }
  }
}

// ---------------------------------------------------------------------------
// Elementwise / reduction kernels
// ---------------------------------------------------------------------------
__global__ void embed_gather_kernel(const int* __restrict__ ids,
                                    const float* __restrict__ emb,
                                    float* __restrict__ X, int rows, int d)
{
  const int row = blockIdx.x;
  const int id  = ids[row];
  for (int i = threadIdx.x; i < d; i += blockDim.x) {
    X[(size_t)row * d + i] = emb[(size_t)id * d + i];
    X[(size_t)rows * d + (size_t)row * d + i] = 0.0f;   // zero tangent
  }
}

// X,H layout [2, rows, d] (primal then tangent)
__global__ void rmsnorm_jvp_kernel(const float* __restrict__ X,
                                   const float* __restrict__ g,
                                   float* __restrict__ H, int rows, int d)
{
  const int row = blockIdx.x, tid = threadIdx.x;
  const float* xp = X + (size_t)row * d;
  const float* xt = X + (size_t)rows * d + (size_t)row * d;
  float s2 = 0.0f, spt = 0.0f;
  for (int i = tid; i < d; i += blockDim.x) {
    float a = xp[i], b = xt[i];
    s2 += a * a; spt += a * b;
  }
  __shared__ float r1[256], r2[256];
  r1[tid] = s2; r2[tid] = spt; __syncthreads();
  for (int s = 128; s > 0; s >>= 1) {
    if (tid < s) { r1[tid] += r1[tid + s]; r2[tid] += r2[tid + s]; }
    __syncthreads();
  }
  const float inv  = rsqrtf(r1[0] / (float)d + EPS_RMS);
  const float c    = r2[0] / (float)d;
  const float inv3 = inv * inv * inv;
  float* hp = H + (size_t)row * d;
  float* ht = H + (size_t)rows * d + (size_t)row * d;
  for (int i = tid; i < d; i += blockDim.x) {
    float gg = g[i], a = xp[i], b = xt[i];
    hp[i] = a * gg * inv;
    ht[i] = gg * (b * inv - a * c * inv3);
  }
}

// SP/ST: [CH, S, S]; row-wise causal softmax JVP.  SP <- p, ST <- dp
__global__ void softmax_jvp_kernel(float* __restrict__ SP, float* __restrict__ ST, int s)
{
  const int q = blockIdx.x, z = blockIdx.y, tid = threadIdx.x;
  float* sp = SP + ((size_t)z * s + q) * s;
  float* st = ST + ((size_t)z * s + q) * s;
  const int nv = q + 1;
  __shared__ float red[256];

  float mx = -1e30f;
  for (int j = tid; j < nv; j += blockDim.x) mx = fmaxf(mx, sp[j]);
  red[tid] = mx; __syncthreads();
  for (int w = 128; w > 0; w >>= 1) { if (tid < w) red[tid] = fmaxf(red[tid], red[tid + w]); __syncthreads(); }
  mx = red[0]; __syncthreads();

  float sum = 0.0f;
  for (int j = tid; j < nv; j += blockDim.x) sum += __expf(sp[j] - mx);
  red[tid] = sum; __syncthreads();
  for (int w = 128; w > 0; w >>= 1) { if (tid < w) red[tid] += red[tid + w]; __syncthreads(); }
  const float inv = 1.0f / red[0]; __syncthreads();

  float dot = 0.0f;
  for (int j = tid; j < nv; j += blockDim.x) dot += __expf(sp[j] - mx) * inv * st[j];
  red[tid] = dot; __syncthreads();
  for (int w = 128; w > 0; w >>= 1) { if (tid < w) red[tid] += red[tid + w]; __syncthreads(); }
  dot = red[0]; __syncthreads();

  for (int j = tid; j < s; j += blockDim.x) {
    if (j < nv) {
      float p = __expf(sp[j] - mx) * inv;
      float d = p * (st[j] - dot);
      sp[j] = p; st[j] = d;
    } else { sp[j] = 0.0f; st[j] = 0.0f; }
  }
}

// G layout [2, n] — tanh-approx GELU JVP, in place
__global__ void gelu_jvp_kernel(float* __restrict__ G, size_t n)
{
  const size_t stride = (size_t)gridDim.x * blockDim.x;
  for (size_t i = (size_t)blockIdx.x * blockDim.x + threadIdx.x; i < n; i += stride) {
    const float x = G[i], dx = G[n + i];
    const float x2 = x * x;
    const float u  = 0.7978845608028654f * (x + 0.044715f * x * x2);
    const float t  = tanhf(u);
    const float du = 0.7978845608028654f * (1.0f + 0.134145f * x2);
    const float dg = 0.5f * (1.0f + t) + 0.5f * x * (1.0f - t * t) * du;
    G[i]     = 0.5f * x * (1.0f + t);
    G[n + i] = dg * dx;
  }
}

__global__ void add_pt_kernel(const float* __restrict__ Hp, const float* __restrict__ Ht,
                              float* __restrict__ Y, size_t n)
{
  const size_t stride = (size_t)gridDim.x * blockDim.x;
  for (size_t i = (size_t)blockIdx.x * blockDim.x + threadIdx.x; i < n; i += stride)
    Y[i] = Hp[i] + Ht[i];
}

// ---------------------------------------------------------------------------
// Host orchestration
// ---------------------------------------------------------------------------
static inline void gemm(hipStream_t st, const float* A, const float* B, float* C,
                        int M, int N, int K, int lda, int ldb, int ldc,
                        long long sAz, long long sBz, long long sCz, int zb,
                        float alpha, int transB, int accumulate)
{
  dim3 g((N + BN - 1) / BN, (M + BM - 1) / BM, zb), b(256);
  gemm_bf16_wmma<<<g, b, 0, st>>>(A, B, C, M, N, K, lda, ldb, ldc,
                                  sAz, sBz, sCz, alpha, transB, accumulate);
}

extern "C" void kernel_launch(void* const* d_in, const int* in_sizes, int n_in,
                              void* d_out, int out_size, void* d_ws, size_t ws_size,
                              hipStream_t stream)
{
  (void)in_sizes; (void)n_in; (void)out_size; (void)ws_size;

  const int*   ids   = (const int*)  d_in[0];
  const float* emb   = (const float*)d_in[1];
  const float* Wq    = (const float*)d_in[2];
  const float* Wk    = (const float*)d_in[3];
  const float* Wv    = (const float*)d_in[4];
  const float* Wo    = (const float*)d_in[5];
  const float* W1    = (const float*)d_in[6];
  const float* W2    = (const float*)d_in[7];
  const float* ln1   = (const float*)d_in[8];
  const float* ln2   = (const float*)d_in[9];
  const float* lnf   = (const float*)d_in[10];
  const float* lm    = (const float*)d_in[11];
  const float* Aq0   = (const float*)d_in[12];
  const float* Av0   = (const float*)d_in[14];
  const float* Bq    = (const float*)d_in[17];   // dBq = Bq (Bq0 == 0)
  const float* Bv    = (const float*)d_in[19];   // dBv = Bv (Bv0 == 0)
  float* out = (float*)d_out;

  // workspace carve-up (floats)
  float* ws = (float*)d_ws;
  size_t o = 0;
  float* X  = ws + o; o += (size_t)2 * ROWS * DD;          // [2,rows,D] residual
  float* Hb = ws + o; o += (size_t)2 * ROWS * DD;          // normed act
  float* Qb = ws + o; o += (size_t)2 * ROWS * DD;
  float* Kb = ws + o; o += (size_t)2 * ROWS * DD;
  float* Vb = ws + o; o += (size_t)2 * ROWS * DD;
  float* Ob = ws + o; o += (size_t)2 * ROWS * DD;          // attn out
  float* Gb = ws + o; o += (size_t)2 * ROWS * FF;          // ffn hidden
  float* SP = ws + o; o += (size_t)ATT_CHUNK * SS * SS;    // scores primal -> p
  float* ST = ws + o; o += (size_t)ATT_CHUNK * SS * SS;    // scores tangent -> dp
  float* Ub = ws + o; o += (size_t)ROWS * RR;              // LoRA intermediate
  float* Yb = ws + o; o += (size_t)ROWS * DD;              // primal+tangent combined

  const size_t TOFF = (size_t)ROWS * DD;                   // tangent-half offset
  const long long SSQ = (long long)SS * SS;

  embed_gather_kernel<<<ROWS, 256, 0, stream>>>(ids, emb, X, ROWS, DD);

  for (int l = 0; l < L_LAYERS; ++l) {
    const float* wq = Wq + (size_t)l * DD * DD;
    const float* wk = Wk + (size_t)l * DD * DD;
    const float* wv = Wv + (size_t)l * DD * DD;
    const float* wo = Wo + (size_t)l * DD * DD;
    const float* w1 = W1 + (size_t)l * DD * FF;
    const float* w2 = W2 + (size_t)l * FF * DD;

    // h = rms(x, ln1) with tangent
    rmsnorm_jvp_kernel<<<ROWS, 256, 0, stream>>>(X, ln1 + (size_t)l * DD, Hb, ROWS, DD);

    // q,k,v (primal+tangent stacked: M = 2*ROWS)
    gemm(stream, Hb, wq, Qb, 2 * ROWS, DD, DD, DD, DD, DD, 0, 0, 0, 1, 1.0f, 0, 0);
    gemm(stream, Hb, wk, Kb, 2 * ROWS, DD, DD, DD, DD, DD, 0, 0, 0, 1, 1.0f, 0, 0);
    gemm(stream, Hb, wv, Vb, 2 * ROWS, DD, DD, DD, DD, DD, 0, 0, 0, 1, 1.0f, 0, 0);

    // LoRA tangent: qt += 2 * (hp @ Aq0^T) @ Bq^T   (only surviving JVP term)
    gemm(stream, Hb, Aq0 + (size_t)l * RR * DD, Ub, ROWS, RR, DD, DD, DD, RR,
         0, 0, 0, 1, 1.0f, 1, 0);
    gemm(stream, Ub, Bq + (size_t)l * DD * RR, Qb + TOFF, ROWS, DD, RR, RR, RR, DD,
         0, 0, 0, 1, LORA_SCALE, 1, 1);
    // vt += 2 * (hp @ Av0^T) @ Bv^T
    gemm(stream, Hb, Av0 + (size_t)l * RR * DD, Ub, ROWS, RR, DD, DD, DD, RR,
         0, 0, 0, 1, 1.0f, 1, 0);
    gemm(stream, Ub, Bv + (size_t)l * DD * RR, Vb + TOFF, ROWS, DD, RR, RR, RR, DD,
         0, 0, 0, 1, LORA_SCALE, 1, 1);

    // attention, chunks of ATT_CHUNK heads (b constant within a chunk)
    const float sc = 0.125f;   // 1/sqrt(HD)
    for (int c = 0; c < (BB * HH) / ATT_CHUNK; ++c) {
      const int b  = (c * ATT_CHUNK) / HH;
      const int h0 = (c * ATT_CHUNK) % HH;
      const size_t base = (size_t)b * SS * DD + (size_t)h0 * HD;
      const float* Qp = Qb + base;        const float* Qt = Qb + TOFF + base;
      const float* Kp = Kb + base;        const float* Kt = Kb + TOFF + base;
      const float* Vp = Vb + base;        const float* Vt = Vb + TOFF + base;
      float* Op = Ob + base;              float* Ot = Ob + TOFF + base;

      // sp = sc * qp kp^T ; st = sc * (qt kp^T + qp kt^T)
      gemm(stream, Qp, Kp, SP, SS, SS, HD, DD, DD, SS, HD, HD, SSQ, ATT_CHUNK, sc, 1, 0);
      gemm(stream, Qt, Kp, ST, SS, SS, HD, DD, DD, SS, HD, HD, SSQ, ATT_CHUNK, sc, 1, 0);
      gemm(stream, Qp, Kt, ST, SS, SS, HD, DD, DD, SS, HD, HD, SSQ, ATT_CHUNK, sc, 1, 1);

      softmax_jvp_kernel<<<dim3(SS, ATT_CHUNK), 256, 0, stream>>>(SP, ST, SS);

      // op = p vp ; ot = dp vp + p vt
      gemm(stream, SP, Vp, Op, SS, HD, SS, SS, DD, DD, SSQ, HD, HD, ATT_CHUNK, 1.0f, 0, 0);
      gemm(stream, ST, Vp, Ot, SS, HD, SS, SS, DD, DD, SSQ, HD, HD, ATT_CHUNK, 1.0f, 0, 0);
      gemm(stream, SP, Vt, Ot, SS, HD, SS, SS, DD, DD, SSQ, HD, HD, ATT_CHUNK, 1.0f, 0, 1);
    }

    // x += o @ Wo   (residual via accumulate, both halves)
    gemm(stream, Ob, wo, X, 2 * ROWS, DD, DD, DD, DD, DD, 0, 0, 0, 1, 1.0f, 0, 1);

    // FFN
    rmsnorm_jvp_kernel<<<ROWS, 256, 0, stream>>>(X, ln2 + (size_t)l * DD, Hb, ROWS, DD);
    gemm(stream, Hb, w1, Gb, 2 * ROWS, FF, DD, DD, FF, FF, 0, 0, 0, 1, 1.0f, 0, 0);
    gelu_jvp_kernel<<<4096, 256, 0, stream>>>(Gb, (size_t)ROWS * FF);
    gemm(stream, Gb, w2, X, 2 * ROWS, DD, FF, FF, DD, DD, 0, 0, 0, 1, 1.0f, 0, 1);
  }

  // logits = (rms_p + rms_t) @ lm_head   (out + 1.0 * dp fused into one GEMM)
  rmsnorm_jvp_kernel<<<ROWS, 256, 0, stream>>>(X, lnf, Hb, ROWS, DD);
  add_pt_kernel<<<2048, 256, 0, stream>>>(Hb, Hb + TOFF, Yb, (size_t)ROWS * DD);
  gemm(stream, Yb, lm, out, ROWS, VV, DD, DD, VV, VV, 0, 0, 0, 1, 1.0f, 0, 0);
}
